// SparsePertokenMoE_1168231105047
// MI455X (gfx1250) — matrix operand order repaired and checked by reading, not compiled, not run on hardware
//
#include <hip/hip_runtime.h>
#include <math.h>

#define DIM   1024
#define HID   4096
#define T_TOK 2048
#define NEXP  8
#define NROUT 7
#define ALPHA 2.0f
#define MT    32      // tokens per block tile (2 WMMA m-tiles)
#define NT    256     // output cols per block tile (16 WMMA n-tiles, 2/wave)

typedef __attribute__((ext_vector_type(16))) __bf16 v16bf;
typedef __attribute__((ext_vector_type(8)))  float  v8f;

static __device__ __forceinline__ __bf16 f2bf(float f) { return (__bf16)f; }
static __device__ __forceinline__ int imin(int a, int b) { return a < b ? a : b; }

static __device__ __forceinline__ unsigned int pack2(float lo, float hi) {
    union { __bf16 h[2]; unsigned int u; } u2;
    u2.h[0] = f2bf(lo);
    u2.h[1] = f2bf(hi);
    return u2.u;   // lowers to v_cvt_pk_bf16_f32
}

// ---------------------------------------------------------------------------
// WMMA fragment layouts (ISA 7.12.2, 16-bit wave32):
// A 16x32: lane = M + 16*((K>>3)&1); half = 2*((K>>4)*4 + ((K&7)>>1)) + (K&1)
// B 32x16: lane = (N&15) + 16*((K>>3)&1); half = 2*((K>>4)*4 + ((K&7)>>1)) + (K&1)
// A pair (K even, K+1) and B pair occupy adjacent 16-bit slots -> pack as b32.
// ---------------------------------------------------------------------------
static __device__ __forceinline__ int a_off(int m, int kk) {
    int alane = m + (((kk >> 3) & 1) << 4);
    int vslot = ((kk >> 4) << 2) + ((kk & 7) >> 1);
    return alane * 16 + vslot * 2 + (kk & 1);
}

// column-per-thread B staging: 32 coalesced b32 loads, 16 cvt_pk, 4 ds_store_b128
static __device__ __forceinline__ void stage_b_col(__bf16* __restrict__ Bsh,
                                                   const float* __restrict__ col,
                                                   int k0, size_t stride,
                                                   int btile, int bl) {
#pragma unroll
    for (int g = 0; g < 2; ++g) {
        unsigned int pk[8];
#pragma unroll
        for (int j = 0; j < 8; ++j) {
            float v0 = col[(size_t)(k0 + g * 8 + 2 * j + ((j >> 2) << 3))     * stride];
            float v1 = col[(size_t)(k0 + g * 8 + 2 * j + ((j >> 2) << 3) + 1) * stride];
            pk[j] = pack2(v0, v1);
        }
        __bf16* dst = Bsh + btile * 512 + (bl + g * 16) * 16;
        *(uint4*)(dst)     = make_uint4(pk[0], pk[1], pk[2], pk[3]);
        *(uint4*)(dst + 8) = make_uint4(pk[4], pk[5], pk[6], pk[7]);
    }
}

// ---------------------------------------------------------------------------
// K0: zero the per-expert counters (d_ws is poisoned by the harness)
// ---------------------------------------------------------------------------
__global__ void zero_counts_kernel(int* __restrict__ counts) {
    if (threadIdx.x < 8) counts[threadIdx.x] = 0;
}

// ---------------------------------------------------------------------------
// K1: router — one wave per token, butterfly reduce, softmax over 8 experts.
// ---------------------------------------------------------------------------
__global__ void __launch_bounds__(256)
router_kernel(const float* __restrict__ x, const float* __restrict__ wr,
              float* __restrict__ scale, int* __restrict__ eid) {
    const int wave = threadIdx.x >> 5;
    const int lane = threadIdx.x & 31;
    const int t = blockIdx.x * 8 + wave;
    if (t >= T_TOK) return;

    float acc[8] = {0.f, 0.f, 0.f, 0.f, 0.f, 0.f, 0.f, 0.f};
    const float* xr = x + (size_t)t * DIM;
    for (int d = lane; d < DIM; d += 32) {
        float xv = xr[d];
        const float4 w0 = *(const float4*)(wr + (size_t)d * 8);
        const float4 w1 = *(const float4*)(wr + (size_t)d * 8 + 4);
        acc[0] += xv * w0.x; acc[1] += xv * w0.y;
        acc[2] += xv * w0.z; acc[3] += xv * w0.w;
        acc[4] += xv * w1.x; acc[5] += xv * w1.y;
        acc[6] += xv * w1.z; acc[7] += xv * w1.w;
    }
#pragma unroll
    for (int e = 0; e < 8; ++e)
#pragma unroll
        for (int m = 16; m > 0; m >>= 1) acc[e] += __shfl_xor(acc[e], m, 32);

    if (lane == 0) {
        float mx = acc[0];
#pragma unroll
        for (int e = 1; e < 8; ++e) mx = fmaxf(mx, acc[e]);
        float s = 0.f, p[8];
#pragma unroll
        for (int e = 0; e < 8; ++e) { p[e] = __expf(acc[e] - mx); s += p[e]; }
        int best = 0;
        float bp = p[0];
#pragma unroll
        for (int e = 1; e < 8; ++e) if (p[e] > bp) { bp = p[e]; best = e; }
        eid[t] = best;
        scale[t] = (best < NROUT) ? ALPHA * (bp / s) : 0.f;
    }
}

// ---------------------------------------------------------------------------
// K2: compaction — per-expert token lists via atomics.
// ---------------------------------------------------------------------------
__global__ void compact_kernel(const int* __restrict__ eid,
                               int* __restrict__ counts,
                               int* __restrict__ lists) {
    const int t = blockIdx.x * blockDim.x + threadIdx.x;
    if (t >= T_TOK) return;
    const int e = eid[t];
    if (e >= 0 && e < NROUT) {
        int pos = atomicAdd(&counts[e], 1);
        lists[e * T_TOK + pos] = t;
    }
}

// ---------------------------------------------------------------------------
// up-gate pass: H = (X @ Wup) * silu(X @ Wgate), bf16 out.
// Block tile: MT=32 tokens x NT=256 hid cols, K = DIM. 8 waves, 8 WMMA/wave/K.
// ROUTED: rows gathered via expert list; H indexed by token id (no collisions).
// ---------------------------------------------------------------------------
template <bool ROUTED>
__global__ void __launch_bounds__(256)
moe_up_gate_kernel(const float* __restrict__ X,
                   const float* __restrict__ WupBase,
                   const float* __restrict__ WgtBase,
                   __bf16* __restrict__ Hout,
                   const int* __restrict__ lists,
                   const int* __restrict__ counts) {
    __shared__ __align__(32) __bf16 Ash[1024];   // 2 A-tiles
    __shared__ __align__(32) __bf16 Bup[8192];   // 16 B-tiles
    __shared__ __align__(32) __bf16 Bgt[8192];
    __shared__ int toks[MT];

    const int tile  = blockIdx.x;
    const int hbase = blockIdx.y * NT;
    const float* Wup = WupBase;
    const float* Wgt = WgtBase;
    const int* list = nullptr;
    int cnt = T_TOK;
    if (ROUTED) {
        const int e = blockIdx.z;
        cnt = counts[e];
        if (tile * MT >= cnt) return;                 // uniform early-exit
        Wup  = WupBase + (size_t)e * DIM * HID;
        Wgt  = WgtBase + (size_t)e * DIM * HID;
        list = lists + e * T_TOK;
    }
    const int tid = threadIdx.x;
    if (tid < MT) {
        int pos = tile * MT + tid;
        toks[tid] = ROUTED ? list[imin(pos, cnt - 1)] : pos;
    }
    __syncthreads();

    // ---- fixed staging geometry, hoisted out of the K loop ----
    const int am0 = tid >> 4;                 // A rows: am0 (0..15) and am0+16
    const int am1 = am0 + 16;
    const int akk = (tid & 15) << 1;          // even K within 32-slab
    const float* xr0 = X + (size_t)toks[am0] * DIM + akk;
    const float* xr1 = X + (size_t)toks[am1] * DIM + akk;
    const int aoff0 = a_off(am0, akk);        // tile 0
    const int aoff1 = 512 + a_off(am1 & 15, akk);

    const int bn    = tid;                    // B column (0..255)
    const int btile = bn >> 4;
    const int bl    = bn & 15;
    const float* wuc = Wup + hbase + bn;
    const float* wgc = Wgt + hbase + bn;

    const int wave = tid >> 5, lane = tid & 31;
    v8f aU[2][2] = {{{}, {}}, {{}, {}}};      // [m-tile][n-tile]
    v8f aG[2][2] = {{{}, {}}, {{}, {}}};

    for (int k0 = 0; k0 < DIM; k0 += 32) {
        // stage A: float2 -> cvt_pk -> b32 store, two row-pairs per thread
        {
            const float2 v0 = *(const float2*)(xr0 + k0);
            const float2 v1 = *(const float2*)(xr1 + k0);
            *(unsigned int*)(Ash + aoff0) = pack2(v0.x, v0.y);
            *(unsigned int*)(Ash + aoff1) = pack2(v1.x, v1.y);
        }
        // stage B slabs (up + gate), column-per-thread
        stage_b_col(Bup, wuc, k0, HID, btile, bl);
        stage_b_col(Bgt, wgc, k0, HID, btile, bl);
        if (k0 + 32 < DIM) {
            __builtin_prefetch(wuc + (size_t)(k0 + 32) * HID, 0, 0);
            __builtin_prefetch(wgc + (size_t)(k0 + 32) * HID, 0, 0);
        }
        __syncthreads();

        const v16bf a0  = *(const v16bf*)(Ash + lane * 16);
        const v16bf a1  = *(const v16bf*)(Ash + 512 + lane * 16);
        const v16bf bu0 = *(const v16bf*)(Bup + (wave * 2    ) * 512 + lane * 16);
        const v16bf bu1 = *(const v16bf*)(Bup + (wave * 2 + 1) * 512 + lane * 16);
        const v16bf bg0 = *(const v16bf*)(Bgt + (wave * 2    ) * 512 + lane * 16);
        const v16bf bg1 = *(const v16bf*)(Bgt + (wave * 2 + 1) * 512 + lane * 16);
        aU[0][0] = __builtin_amdgcn_wmma_f32_16x16x32_bf16(false, a0, false, bu0, (short)0, aU[0][0], false, false);
        aU[0][1] = __builtin_amdgcn_wmma_f32_16x16x32_bf16(false, a0, false, bu1, (short)0, aU[0][1], false, false);
        aU[1][0] = __builtin_amdgcn_wmma_f32_16x16x32_bf16(false, a1, false, bu0, (short)0, aU[1][0], false, false);
        aU[1][1] = __builtin_amdgcn_wmma_f32_16x16x32_bf16(false, a1, false, bu1, (short)0, aU[1][1], false, false);
        aG[0][0] = __builtin_amdgcn_wmma_f32_16x16x32_bf16(false, a0, false, bg0, (short)0, aG[0][0], false, false);
        aG[0][1] = __builtin_amdgcn_wmma_f32_16x16x32_bf16(false, a0, false, bg1, (short)0, aG[0][1], false, false);
        aG[1][0] = __builtin_amdgcn_wmma_f32_16x16x32_bf16(false, a1, false, bg0, (short)0, aG[1][0], false, false);
        aG[1][1] = __builtin_amdgcn_wmma_f32_16x16x32_bf16(false, a1, false, bg1, (short)0, aG[1][1], false, false);
        __syncthreads();
    }

    // C/D layout: VGPR r -> M = r + 8*(lane>=16), N = lane & 15
    const int mrb  = (lane >> 4) << 3;
    const int ncol = lane & 15;
#pragma unroll
    for (int mt = 0; mt < 2; ++mt) {
#pragma unroll
        for (int nt = 0; nt < 2; ++nt) {
            const size_t col = hbase + (wave * 2 + nt) * 16 + ncol;
#pragma unroll
            for (int r = 0; r < 8; ++r) {
                const int m = mt * 16 + mrb + r;
                const int hrow = ROUTED ? toks[m] : (tile * MT + m);
                const float u = aU[mt][nt][r], g = aG[mt][nt][r];
                const float h = u * g / (1.f + __expf(-g));
                Hout[(size_t)hrow * HID + col] = f2bf(h);
            }
        }
    }
}

// ---------------------------------------------------------------------------
// down pass: Out = H @ Wdown  (shared: write; routed: += scale[tok] * ..)
// ---------------------------------------------------------------------------
template <bool ROUTED>
__global__ void __launch_bounds__(256)
moe_down_kernel(const __bf16* __restrict__ H,
                const float* __restrict__ WdnBase,
                float* __restrict__ Out,
                const int* __restrict__ lists,
                const int* __restrict__ counts,
                const float* __restrict__ scale) {
    __shared__ __align__(32) __bf16 Ash[1024];
    __shared__ __align__(32) __bf16 Bdn[8192];
    __shared__ int toks[MT];

    const int tile  = blockIdx.x;
    const int nbase = blockIdx.y * NT;
    const float* Wdn = WdnBase;
    const int* list = nullptr;
    int cnt = T_TOK;
    if (ROUTED) {
        const int e = blockIdx.z;
        cnt = counts[e];
        if (tile * MT >= cnt) return;
        Wdn  = WdnBase + (size_t)e * HID * DIM;
        list = lists + e * T_TOK;
    }
    const int tid = threadIdx.x;
    if (tid < MT) {
        int pos = tile * MT + tid;
        toks[tid] = ROUTED ? list[imin(pos, cnt - 1)] : pos;
    }
    __syncthreads();

    // ---- fixed staging geometry ----
    const int am0 = tid >> 4;
    const int am1 = am0 + 16;
    const int akk = (tid & 15) << 1;
    const int hrow0 = ROUTED ? toks[am0] : (tile * MT + am0);
    const int hrow1 = ROUTED ? toks[am1] : (tile * MT + am1);
    const __bf16* hr0 = H + (size_t)hrow0 * HID + akk;
    const __bf16* hr1 = H + (size_t)hrow1 * HID + akk;
    const int aoff0 = a_off(am0, akk);
    const int aoff1 = 512 + a_off(am1 & 15, akk);

    const int bn    = tid;
    const int btile = bn >> 4;
    const int bl    = bn & 15;
    const float* wdc = Wdn + nbase + bn;

    const int wave = tid >> 5, lane = tid & 31;
    v8f acc[2][2] = {{{}, {}}, {{}, {}}};

    for (int k0 = 0; k0 < HID; k0 += 32) {
        // stage A from bf16 H: raw b32 pair copy (already packed)
        *(unsigned int*)(Ash + aoff0) = *(const unsigned int*)(hr0 + k0);
        *(unsigned int*)(Ash + aoff1) = *(const unsigned int*)(hr1 + k0);
        // stage B slab from fp32 Wdown
        stage_b_col(Bdn, wdc, k0, DIM, btile, bl);
        if (k0 + 32 < HID) __builtin_prefetch(wdc + (size_t)(k0 + 32) * DIM, 0, 0);
        __syncthreads();

        const v16bf a0 = *(const v16bf*)(Ash + lane * 16);
        const v16bf a1 = *(const v16bf*)(Ash + 512 + lane * 16);
        const v16bf b0 = *(const v16bf*)(Bdn + (wave * 2    ) * 512 + lane * 16);
        const v16bf b1 = *(const v16bf*)(Bdn + (wave * 2 + 1) * 512 + lane * 16);
        acc[0][0] = __builtin_amdgcn_wmma_f32_16x16x32_bf16(false, a0, false, b0, (short)0, acc[0][0], false, false);
        acc[0][1] = __builtin_amdgcn_wmma_f32_16x16x32_bf16(false, a0, false, b1, (short)0, acc[0][1], false, false);
        acc[1][0] = __builtin_amdgcn_wmma_f32_16x16x32_bf16(false, a1, false, b0, (short)0, acc[1][0], false, false);
        acc[1][1] = __builtin_amdgcn_wmma_f32_16x16x32_bf16(false, a1, false, b1, (short)0, acc[1][1], false, false);
        __syncthreads();
    }

    const int mrb  = (lane >> 4) << 3;
    const int ncol = lane & 15;
#pragma unroll
    for (int mt = 0; mt < 2; ++mt) {
#pragma unroll
        for (int nt = 0; nt < 2; ++nt) {
            const size_t col = nbase + (wave * 2 + nt) * 16 + ncol;
#pragma unroll
            for (int r = 0; r < 8; ++r) {
                const int m = mt * 16 + mrb + r;
                const int pos = tile * MT + m;
                if (ROUTED) {
                    if (pos < cnt) {
                        const int tok = toks[m];
                        Out[(size_t)tok * DIM + col] += scale[tok] * acc[mt][nt][r];
                    }
                } else {
                    Out[(size_t)pos * DIM + col] = acc[mt][nt][r];
                }
            }
        }
    }
}

// ---------------------------------------------------------------------------
// Host launcher
// ---------------------------------------------------------------------------
extern "C" void kernel_launch(void* const* d_in, const int* in_sizes, int n_in,
                              void* d_out, int out_size, void* d_ws, size_t ws_size,
                              hipStream_t stream) {
    const float* x        = (const float*)d_in[0];
    const float* w_router = (const float*)d_in[1];
    const float* w_up     = (const float*)d_in[2];
    const float* w_gate   = (const float*)d_in[3];
    const float* w_down   = (const float*)d_in[4];
    const float* ws_up    = (const float*)d_in[5];
    const float* ws_gate  = (const float*)d_in[6];
    const float* ws_down  = (const float*)d_in[7];
    float* out = (float*)d_out;

    // workspace layout
    char* ws = (char*)d_ws;
    __bf16* H = (__bf16*)ws;                               // T*HID bf16 = 16 MB
    size_t off = (size_t)T_TOK * HID * sizeof(__bf16);
    float* scale = (float*)(ws + off); off += (size_t)T_TOK * sizeof(float);
    int* eid     = (int*)(ws + off);   off += (size_t)T_TOK * sizeof(int);
    int* counts  = (int*)(ws + off);   off += 32;
    int* lists   = (int*)(ws + off);   off += (size_t)NROUT * T_TOK * sizeof(int);

    zero_counts_kernel<<<1, 32, 0, stream>>>(counts);
    router_kernel<<<T_TOK / 8, 256, 0, stream>>>(x, w_router, scale, eid);
    compact_kernel<<<T_TOK / 256, 256, 0, stream>>>(eid, counts, lists);

    // shared expert: H = swiglu(x); out = H @ ws_down (writes all of d_out)
    moe_up_gate_kernel<false><<<dim3(T_TOK / MT, HID / NT), 256, 0, stream>>>(
        x, ws_up, ws_gate, H, nullptr, nullptr);
    moe_down_kernel<false><<<dim3(T_TOK / MT, DIM / NT), 256, 0, stream>>>(
        H, ws_down, out, nullptr, nullptr, nullptr);

    // routed top-1 expert (grouped GEMM over per-expert token lists)
    moe_up_gate_kernel<true><<<dim3(T_TOK / MT, HID / NT, NROUT), 256, 0, stream>>>(
        x, w_up, w_gate, H, lists, counts);
    moe_down_kernel<true><<<dim3(T_TOK / MT, DIM / NT, NROUT), 256, 0, stream>>>(
        H, w_down, out, lists, counts, scale);

    (void)in_sizes; (void)n_in; (void)out_size; (void)ws_size;
}
